// PortOpt_DL_DeepSig_61761629716514
// MI455X (gfx1250) — matrix operand order repaired and checked by reading, not compiled, not run on hardware
//
#include <hip/hip_runtime.h>

typedef __attribute__((ext_vector_type(16))) __bf16 v16bf;
typedef __attribute__((ext_vector_type(8)))  float  v8f;
typedef unsigned int u32;
typedef __attribute__((ext_vector_type(4)))  u32    u32x4;
typedef unsigned short u16;

union V16 { v16bf bf; u32x4 q[2]; };

__device__ __forceinline__ u16 f2bf(float f) {
  u32 u = __builtin_bit_cast(u32, f);
  u += 0x7FFFu + ((u >> 16) & 1u);           // round-to-nearest-even
  return (u16)(u >> 16);
}

// ---------------- constants ----------------
// B=128, L=256, CIN=8
// stage1 augment: L1OUT=252, C1=14 (8 orig + 1 time + 5 conv)
// sig1: 251 steps, 2954 channels, padded row stride 2960 (16B-aligned bf16 rows)
// stage2 conv: GEMM M=128*247=31616, N=64, Kpad=463*32=14816 (k = j*2960 + c)
// sig2: C2=9 -> 819 channels
#define SIG1_STRIDE 2960
#define KTILES 463
#define MROWS 31616

// ---------------- 1: BatchNorm over (B,C) per L ----------------
__global__ __launch_bounds__(256) void bn_kernel(const float* __restrict__ x,
    const float* __restrict__ g, const float* __restrict__ bt, float* __restrict__ xn) {
  __shared__ float ss[256], sq[256];
  const int l = blockIdx.x, tid = threadIdx.x;
  float vals[4], s = 0.f, q = 0.f;
  for (int r = 0; r < 4; ++r) {
    int e = tid + r * 256;
    int b = e >> 3, c = e & 7;
    float v = x[b * 2048 + l * 8 + c];
    vals[r] = v; s += v; q += v * v;
  }
  ss[tid] = s; sq[tid] = q;
  __syncthreads();
  for (int off = 128; off > 0; off >>= 1) {
    if (tid < off) { ss[tid] += ss[tid + off]; sq[tid] += sq[tid + off]; }
    __syncthreads();
  }
  float mean = ss[0] * (1.f / 1024.f);
  float var  = sq[0] * (1.f / 1024.f) - mean * mean;
  float sc = g[l] * rsqrtf(var + 1e-5f);
  float sh = bt[l] - mean * sc;
  for (int r = 0; r < 4; ++r) {
    int e = tid + r * 256;
    int b = e >> 3, c = e & 7;
    xn[b * 2048 + l * 8 + c] = vals[r] * sc + sh;
  }
}

// ---------------- 2: Augment1 -> (B,252,14) ----------------
__global__ __launch_bounds__(256) void aug1_kernel(const float* __restrict__ xn,
    const float* __restrict__ w1, const float* __restrict__ b1,
    const float* __restrict__ w2, const float* __restrict__ b2,
    float* __restrict__ s1) {
  int idx = blockIdx.x * 256 + threadIdx.x;   // 128*252 exactly
  int b = idx / 252, t = idx % 252;
  const float* xb = xn + b * 2048 + t * 8;
  float mid[10];
  for (int o = 0; o < 10; ++o) mid[o] = b1[o];
  for (int j = 0; j < 5; ++j)
    for (int c = 0; c < 8; ++c) {
      float xv = xb[j * 8 + c];
      for (int o = 0; o < 10; ++o) mid[o] += xv * w1[o * 40 + c * 5 + j];
    }
  float* out = s1 + (b * 252 + t) * 14;
  for (int c = 0; c < 8; ++c) out[c] = xb[32 + c];     // original, truncated by K-1=4
  out[8] = (float)t * (1.f / 251.f);                   // time channel
  for (int o2 = 0; o2 < 5; ++o2) {
    float a = b2[o2];
    for (int o = 0; o < 10; ++o) a += fmaxf(mid[o], 0.f) * w2[o2 * 10 + o];
    out[9 + o2] = a;
  }
}

// ---------------- 3: depth-3 streamed signature, C=14, in LDS ----------------
__global__ __launch_bounds__(256) void sig1_kernel(const float* __restrict__ s1,
                                                   u16* __restrict__ sigb) {
  __shared__ float L1[14], L2[196], L3[2744], V[14];
  const int b = blockIdx.x, tid = threadIdx.x;
  for (int e = tid; e < 2744; e += 256) L3[e] = 0.f;
  if (tid < 196) L2[tid] = 0.f;
  if (tid < 14)  L1[tid] = 0.f;
  __syncthreads();
  const float* sb = s1 + b * 252 * 14;
  for (int s = 0; s < 251; ++s) {
    if (tid < 14) V[tid] = sb[(s + 1) * 14 + tid] - sb[s * 14 + tid];
    __syncthreads();
    // Chen: new3 = a3 + v^3/6 + a1 (x) v^2/2 + a2 (x) v   (reads old a1,a2)
    for (int e = tid; e < 2744; e += 256) {
      int i = e / 196, j = (e / 14) % 14, k = e % 14;
      L3[e] += V[i] * V[j] * V[k] * (1.f / 6.f)
             + 0.5f * L1[i] * V[j] * V[k]
             + L2[i * 14 + j] * V[k];
    }
    __syncthreads();
    if (tid < 196) {
      int i = tid / 14, j = tid % 14;
      L2[tid] += 0.5f * V[i] * V[j] + L1[i] * V[j];
    }
    __syncthreads();
    if (tid < 14) L1[tid] += V[tid];
    __syncthreads();
    u16* o = sigb + (long long)(b * 251 + s) * SIG1_STRIDE;
    for (int e = tid; e < SIG1_STRIDE; e += 256) {
      float v = (e < 14) ? L1[e] : (e < 210) ? L2[e - 14] : (e < 2954) ? L3[e - 210] : 0.f;
      o[e] = f2bf(v);
    }
    __syncthreads();
  }
}

// ---------------- 4: pack a2_w1 into WMMA B-fragment layout (bf16) ----------------
// element (kt, nt, lane, i): k = kt*32 + 8*(lane>=16) + (i&7) + 16*(i>=8) ; n = nt*16 + (lane&15)
__global__ __launch_bounds__(256) void pack_kernel(const float* __restrict__ w,
                                                   u16* __restrict__ bp) {
  int idx = blockIdx.x * 256 + threadIdx.x;   // 463*2048 exactly
  int i    = idx & 15;
  int lane = (idx >> 4) & 31;
  int nt   = (idx >> 9) & 3;
  int kt   = idx >> 11;
  int k = kt * 32 + ((lane >> 4) << 3) + (i & 7) + ((i >> 3) << 4);
  int n = nt * 16 + (lane & 15);
  int j = k / SIG1_STRIDE, c = k % SIG1_STRIDE;
  float v = (j < 5 && c < 2954) ? w[n * 14770 + c * 5 + j] : 0.f;   // (64,2954,5)
  bp[idx] = f2bf(v);
}

// ---------------- 5: bf16 WMMA GEMM with async-LDS double-buffered B ----------------
// (31616 x 14816) x (14816 x 64); 4 waves/block share each 4KB B tile via LDS.
__global__ __launch_bounds__(128) void gemm_kernel(const u16* __restrict__ sigb,
    const u16* __restrict__ bp, float* __restrict__ out) {
  __shared__ __attribute__((aligned(16))) u16 bsm[2][2048];   // double-buffered B tile
  const int tid = threadIdx.x;
  const int wave = tid >> 5, lane = tid & 31;
  const int m0  = blockIdx.x * 64 + wave * 16;
  const int row = m0 + (lane & 15);
  const int bb = row / 247, t = row % 247;
  // A fragment: lane<16 holds K {0..7,16..23}; lane>=16 holds K {8..15,24..31}
  const u16* pA = sigb + (long long)(bb * 251 + t) * SIG1_STRIDE + ((lane >> 4) << 3);
  v8f c0 = {}, c1 = {}, c2 = {}, c3 = {};

  // prologue: async-stage B tile kt=0 into buffer 0 (each thread DMAs 32B)
  {
    const u16* src = bp + tid * 16;
    u32 lo = (u32)(uintptr_t)(&bsm[0][tid * 16]);
    asm volatile("global_load_async_to_lds_b128 %0, %1, off"
                 :: "v"(lo), "v"(src) : "memory");
    asm volatile("global_load_async_to_lds_b128 %0, %1, off offset:16"
                 :: "v"(lo), "v"(src) : "memory");
    asm volatile("s_wait_asynccnt 0" ::: "memory");
  }
  __syncthreads();

  for (int kt = 0; kt < KTILES; ++kt) {
    const int cur = kt & 1;
    // kick off DMA of next B tile into the other buffer (overlaps WMMAs below)
    if (kt + 1 < KTILES) {
      const u16* src = bp + (size_t)(kt + 1) * 2048 + tid * 16;
      u32 lo = (u32)(uintptr_t)(&bsm[cur ^ 1][tid * 16]);
      asm volatile("global_load_async_to_lds_b128 %0, %1, off"
                   :: "v"(lo), "v"(src) : "memory");
      asm volatile("global_load_async_to_lds_b128 %0, %1, off offset:16"
                   :: "v"(lo), "v"(src) : "memory");
    }
    V16 a, b0, b1, b2, b3;
    const u16* pa = pA + kt * 32;
    a.q[0] = *(const u32x4*)(pa);
    a.q[1] = *(const u32x4*)(pa + 16);
    __builtin_prefetch((const void*)(pa + 64), 0, 3);   // A tile kt+2 -> global_prefetch
    const u16* lb = &bsm[cur][lane * 16];
    b0.q[0] = *(const u32x4*)(lb);          b0.q[1] = *(const u32x4*)(lb + 8);
    b1.q[0] = *(const u32x4*)(lb + 512);    b1.q[1] = *(const u32x4*)(lb + 520);
    b2.q[0] = *(const u32x4*)(lb + 1024);   b2.q[1] = *(const u32x4*)(lb + 1032);
    b3.q[0] = *(const u32x4*)(lb + 1536);   b3.q[1] = *(const u32x4*)(lb + 1544);
    c0 = __builtin_amdgcn_wmma_f32_16x16x32_bf16(false, a.bf, false, b0.bf, (short)0, c0, false, false);
    c1 = __builtin_amdgcn_wmma_f32_16x16x32_bf16(false, a.bf, false, b1.bf, (short)0, c1, false, false);
    c2 = __builtin_amdgcn_wmma_f32_16x16x32_bf16(false, a.bf, false, b2.bf, (short)0, c2, false, false);
    c3 = __builtin_amdgcn_wmma_f32_16x16x32_bf16(false, a.bf, false, b3.bf, (short)0, c3, false, false);
    // my DMA for the next tile has landed; barrier also fences everyone's reads
    asm volatile("s_wait_asynccnt 0" ::: "memory");
    __syncthreads();
  }
  // D layout: vgpr r, lanes<16 -> M=r ; lanes>=16 -> M=r+8 ; N=lane&15
  const int mr = m0 + ((lane >> 4) << 3);
  const int n  = lane & 15;
  for (int r = 0; r < 8; ++r) {
    float* o = out + (long long)(mr + r) * 64 + n;
    o[0] = c0[r]; o[16] = c1[r]; o[32] = c2[r]; o[48] = c3[r];
  }
}

// ---------------- 6: bias + ReLU + 1x1 conv + time -> (B,247,9) ----------------
__global__ __launch_bounds__(256) void stream2_kernel(const float* __restrict__ conv1,
    const float* __restrict__ b1, const float* __restrict__ w2,
    const float* __restrict__ b2, float* __restrict__ s2) {
  int idx = blockIdx.x * 256 + threadIdx.x;
  if (idx >= MROWS) return;
  int t = idx % 247;
  float acc[8];
  for (int o2 = 0; o2 < 8; ++o2) acc[o2] = b2[o2];
  const float* cv = conv1 + (long long)idx * 64;
  for (int o = 0; o < 64; ++o) {
    float r = fmaxf(cv[o] + b1[o], 0.f);
    for (int o2 = 0; o2 < 8; ++o2) acc[o2] += r * w2[o2 * 64 + o];
  }
  float* o = s2 + idx * 9;
  o[0] = (float)t * (1.f / 246.f);
  for (int o2 = 0; o2 < 8; ++o2) o[1 + o2] = acc[o2];
}

// ---------------- 7: depth-3 signature, C=9, stream=False ----------------
__global__ __launch_bounds__(256) void sig2_kernel(const float* __restrict__ s2,
                                                   float* __restrict__ sg) {
  __shared__ float L1[9], L2[81], L3[729], V[9];
  const int b = blockIdx.x, tid = threadIdx.x;
  for (int e = tid; e < 729; e += 256) L3[e] = 0.f;
  if (tid < 81) L2[tid] = 0.f;
  if (tid < 9)  L1[tid] = 0.f;
  __syncthreads();
  const float* sb = s2 + b * 247 * 9;
  for (int s = 0; s < 246; ++s) {
    if (tid < 9) V[tid] = sb[(s + 1) * 9 + tid] - sb[s * 9 + tid];
    __syncthreads();
    for (int e = tid; e < 729; e += 256) {
      int i = e / 81, j = (e / 9) % 9, k = e % 9;
      L3[e] += V[i] * V[j] * V[k] * (1.f / 6.f)
             + 0.5f * L1[i] * V[j] * V[k]
             + L2[i * 9 + j] * V[k];
    }
    __syncthreads();
    if (tid < 81) {
      int i = tid / 9, j = tid % 9;
      L2[tid] += 0.5f * V[i] * V[j] + L1[i] * V[j];
    }
    __syncthreads();
    if (tid < 9) L1[tid] += V[tid];
    __syncthreads();
  }
  float* o = sg + b * 819;
  for (int e = tid; e < 819; e += 256)
    o[e] = (e < 9) ? L1[e] : (e < 90) ? L2[e - 9] : L3[e - 90];
}

// ---------------- 8: linear 819->50 + softmax ----------------
__global__ __launch_bounds__(64) void head_kernel(const float* __restrict__ sg,
    const float* __restrict__ w, const float* __restrict__ bias,
    float* __restrict__ out) {
  __shared__ float lg[50];
  __shared__ float mx, sm;
  const int b = blockIdx.x, tid = threadIdx.x;
  const float* s = sg + b * 819;
  if (tid < 50) {
    float a = bias[tid];
    const float* wr = w + tid * 819;
    for (int k = 0; k < 819; ++k) a += s[k] * wr[k];
    lg[tid] = a;
  }
  __syncthreads();
  if (tid == 0) {
    float m = lg[0];
    for (int i = 1; i < 50; ++i) m = fmaxf(m, lg[i]);
    mx = m;
  }
  __syncthreads();
  if (tid < 50) lg[tid] = __expf(lg[tid] - mx);
  __syncthreads();
  if (tid == 0) {
    float t = 0.f;
    for (int i = 0; i < 50; ++i) t += lg[i];
    sm = t;
  }
  __syncthreads();
  if (tid < 50) out[b * 50 + tid] = lg[tid] / sm;
}

extern "C" void kernel_launch(void* const* d_in, const int* in_sizes, int n_in,
                              void* d_out, int out_size, void* d_ws, size_t ws_size,
                              hipStream_t stream) {
  const float* x    = (const float*)d_in[0];
  const float* bng  = (const float*)d_in[1];
  const float* bnb  = (const float*)d_in[2];
  const float* a1w1 = (const float*)d_in[3];
  const float* a1b1 = (const float*)d_in[4];
  const float* a1w2 = (const float*)d_in[5];
  const float* a1b2 = (const float*)d_in[6];
  const float* a2w1 = (const float*)d_in[7];
  const float* a2b1 = (const float*)d_in[8];
  const float* a2w2 = (const float*)d_in[9];
  const float* a2b2 = (const float*)d_in[10];
  const float* linw = (const float*)d_in[11];
  const float* linb = (const float*)d_in[12];
  float* out = (float*)d_out;

  char* ws = (char*)d_ws;
  size_t cur = 0;
  auto carve = [&](size_t bytes) -> char* {
    char* p = ws + cur;
    cur = (cur + bytes + 255) & ~(size_t)255;
    return p;
  };
  float* xn    = (float*)carve((size_t)128 * 256 * 8 * 4);           // 1.0 MB
  float* s1    = (float*)carve((size_t)128 * 252 * 14 * 4);          // 1.8 MB
  float* conv1 = (float*)carve((size_t)MROWS * 64 * 4);              // 8.1 MB
  float* s2    = (float*)carve((size_t)MROWS * 9 * 4);               // 1.1 MB
  float* sg2   = (float*)carve((size_t)128 * 819 * 4);               // 0.4 MB
  u16*   bpack = (u16*)  carve((size_t)KTILES * 2048 * 2);           // 1.9 MB
  u16*   sigb  = (u16*)  carve((size_t)128 * 251 * SIG1_STRIDE * 2 + 64); // ~190 MB (+tail pad)

  bn_kernel     <<<256, 256, 0, stream>>>(x, bng, bnb, xn);
  aug1_kernel   <<<126, 256, 0, stream>>>(xn, a1w1, a1b1, a1w2, a1b2, s1);
  sig1_kernel   <<<128, 256, 0, stream>>>(s1, sigb);
  pack_kernel   <<<3704, 256, 0, stream>>>(a2w1, bpack);
  gemm_kernel   <<<MROWS / 64, 128, 0, stream>>>(sigb, bpack, conv1);
  stream2_kernel<<<(MROWS + 255) / 256, 256, 0, stream>>>(conv1, a2b1, a2w2, a2b2, s2);
  sig2_kernel   <<<128, 256, 0, stream>>>(s2, sg2);
  head_kernel   <<<128, 64, 0, stream>>>(sg2, linw, linb, out);
}
